// Model_78451872628849
// MI455X (gfx1250) — compile-verified, matrix operand order checked by reading, not run on hardware
//
#include <hip/hip_runtime.h>
#include <hip/hip_bf16.h>
#include <math.h>

// Problem dims (compile-time, match reference)
constexpr int Bd = 32;    // batch
constexpr int Td = 512;   // time
constexpr int Id = 128;   // input size
constexpr int Hd = 512;   // hidden size
constexpr int G4H = 4 * Hd; // 2048 gate rows
constexpr int NS = 8;       // j-slices for parallel weighted-history reduction
constexpr int CHUNK_J = 8;  // c_hist rows per LDS staging chunk

typedef __bf16 bf16x16 __attribute__((ext_vector_type(16)));
typedef __bf16 bf16x8  __attribute__((ext_vector_type(8)));
typedef float  f32x8   __attribute__((ext_vector_type(8)));
typedef int    v4i     __attribute__((ext_vector_type(4)));

union F8 { f32x8 v; float f[8]; };

// CDNA5 async global->LDS path (ASYNCcnt-tracked), with sync fallback
#if defined(__has_builtin)
#if __has_builtin(__builtin_amdgcn_global_load_async_to_lds_b128) && \
    __has_builtin(__builtin_amdgcn_s_wait_asynccnt)
#define USE_ASYNC_LDS 1
#endif
#endif

typedef __attribute__((address_space(1))) v4i g_v4i;
typedef __attribute__((address_space(3))) v4i l_v4i;

// Load a 16x32 (A) or 32x16-as-rows-of-Wt (B) bf16 WMMA fragment.
// p points at row_start + k0 + hi*8 ; chunks at p[0..7] and p[16..23]
__device__ __forceinline__ bf16x16 load_frag(const __bf16* p) {
    bf16x8 lo = *(const bf16x8*)(p);
    bf16x8 hi = *(const bf16x8*)(p + 16);
    return __builtin_shufflevector(lo, hi, 0,1,2,3,4,5,6,7,8,9,10,11,12,13,14,15);
}

__device__ __forceinline__ float sigmoidf_(float x) {
    return 1.0f / (1.0f + __expf(-x));
}

// ---------------------------------------------------------------------------
// Prep: bf16 conversions of weights/values, bias_sum, zero Cpart / h0 buffers
// ---------------------------------------------------------------------------
__global__ void prep_kernel(const float* __restrict__ values,
                            const float* __restrict__ W_ih,
                            const float* __restrict__ W_hh,
                            const float* __restrict__ b_ih,
                            const float* __restrict__ b_hh,
                            const float* __restrict__ atten_W,
                            __bf16* __restrict__ values_bf,
                            __bf16* __restrict__ wih_bf,
                            __bf16* __restrict__ whh_bf,
                            __bf16* __restrict__ aw_bf,
                            float*  __restrict__ bias_sum,
                            float*  __restrict__ Cpart,
                            __bf16* __restrict__ hbuf0) {
    size_t tid = (size_t)blockIdx.x * blockDim.x + threadIdx.x;
    size_t nt  = (size_t)gridDim.x * blockDim.x;
    for (size_t i = tid; i < (size_t)Bd * Td * Id; i += nt) values_bf[i] = (__bf16)values[i];
    for (size_t i = tid; i < (size_t)G4H * Id;     i += nt) wih_bf[i]    = (__bf16)W_ih[i];
    for (size_t i = tid; i < (size_t)G4H * Hd;     i += nt) whh_bf[i]    = (__bf16)W_hh[i];
    for (size_t i = tid; i < (size_t)Id * Id;      i += nt) aw_bf[i]     = (__bf16)atten_W[i];
    for (size_t i = tid; i < (size_t)G4H;          i += nt) bias_sum[i]  = b_ih[i] + b_hh[i];
    for (size_t i = tid; i < (size_t)NS * Bd * Hd; i += nt) Cpart[i]     = 0.0f;
    for (size_t i = tid; i < (size_t)Bd * Hd;      i += nt) hbuf0[i]     = (__bf16)0.0f;
}

// ---------------------------------------------------------------------------
// Cumulative sum of Deltas along T -> dcum[B,T]
// ---------------------------------------------------------------------------
__global__ void cumsum_kernel(const float* __restrict__ Deltas,
                              float* __restrict__ dcum) {
    int b = blockIdx.x * blockDim.x + threadIdx.x;
    if (b >= Bd) return;
    float s = 0.0f;
    for (int t = 0; t < Td; ++t) {
        s += Deltas[(size_t)b * Td + t];
        dcum[(size_t)b * Td + t] = s;
    }
}

// ---------------------------------------------------------------------------
// v = values @ atten_W^T + atten_b  (WMMA bf16, M=B*T=16384, N=K=128)
// grid: (B*T/16) blocks x 256 threads (8 waves, one 16x16 tile each)
// ---------------------------------------------------------------------------
__global__ void atten_gemm_kernel(const __bf16* __restrict__ vals_bf,
                                  const __bf16* __restrict__ aw_bf,
                                  const float*  __restrict__ atten_b,
                                  float*  __restrict__ vlin,
                                  __bf16* __restrict__ vlin_bf) {
    int w    = threadIdx.x >> 5;
    int lane = threadIdx.x & 31;
    int l    = lane & 15;
    int hi   = lane >> 4;
    int m0   = blockIdx.x * 16;
    int n0   = w * 16;

    F8 acc;
#pragma unroll
    for (int e = 0; e < 8; ++e) acc.f[e] = 0.0f;

    const __bf16* ap = vals_bf + ((size_t)(m0 + l)) * Id + hi * 8;
    const __bf16* bp = aw_bf   + ((size_t)(n0 + l)) * Id + hi * 8;
#pragma unroll
    for (int k0 = 0; k0 < Id; k0 += 32) {
        bf16x16 a = load_frag(ap + k0);
        bf16x16 b = load_frag(bp + k0);
        acc.v = __builtin_amdgcn_wmma_f32_16x16x32_bf16(
            false, a, false, b, (short)0, acc.v, false, false);
    }
    int   n  = n0 + l;
    float bb = atten_b[n];
#pragma unroll
    for (int e = 0; e < 8; ++e) {
        int    m = m0 + hi * 8 + e;
        float  r = acc.f[e] + bb;
        size_t o = (size_t)m * Id + n;
        vlin[o]    = r;
        vlin_bf[o] = (__bf16)r;
    }
}

// ---------------------------------------------------------------------------
// Per-step softmax weights: scores a_tj = v[b,t]·values[b,j] (j<t), masked
// softmax, times decay 1/log(e + dcum[t]-dcum[j])  -> alpha[b, j]
// grid: B blocks x 256 threads
// ---------------------------------------------------------------------------
__global__ void alpha_kernel(const float* __restrict__ values,
                             const float* __restrict__ vlin,
                             const float* __restrict__ dcum,
                             float* __restrict__ alpha,
                             int t) {
    __shared__ float s_v[Id];
    __shared__ float s_red[256];
    int b   = blockIdx.x;
    int tid = threadIdx.x;

    if (tid < Id) s_v[tid] = vlin[((size_t)b * Td + t) * Id + tid];
    __syncthreads();

    float sc[2];
    float lmax = -INFINITY;
#pragma unroll
    for (int r = 0; r < 2; ++r) {
        int j = tid + r * 256;
        sc[r] = -INFINITY;
        if (j < t) {
            const float4* vp = (const float4*)&values[((size_t)b * Td + j) * Id];
            const float4* sv = (const float4*)s_v;
            float acc = 0.0f;
#pragma unroll
            for (int k = 0; k < Id / 4; ++k) {
                float4 a = sv[k], c = vp[k];
                acc += a.x * c.x + a.y * c.y + a.z * c.z + a.w * c.w;
            }
            sc[r] = acc;
            lmax = fmaxf(lmax, acc);
        }
    }
    s_red[tid] = lmax;
    __syncthreads();
    for (int s = 128; s > 0; s >>= 1) {
        if (tid < s) s_red[tid] = fmaxf(s_red[tid], s_red[tid + s]);
        __syncthreads();
    }
    float gmax = s_red[0];
    __syncthreads();

    float ex[2];
    float lsum = 0.0f;
#pragma unroll
    for (int r = 0; r < 2; ++r) {
        int j = tid + r * 256;
        ex[r] = 0.0f;
        if (j < t) { ex[r] = __expf(sc[r] - gmax); lsum += ex[r]; }
    }
    s_red[tid] = lsum;
    __syncthreads();
    for (int s = 128; s > 0; s >>= 1) {
        if (tid < s) s_red[tid] += s_red[tid + s];
        __syncthreads();
    }
    float inv = 1.0f / s_red[0];
    float dct = dcum[(size_t)b * Td + t];
#pragma unroll
    for (int r = 0; r < 2; ++r) {
        int j = tid + r * 256;
        if (j < t) {
            float w = 1.0f / __logf(2.7182818284590452f + dct - dcum[(size_t)b * Td + j]);
            alpha[(size_t)b * Td + j] = ex[r] * inv * w;
        }
    }
}

// ---------------------------------------------------------------------------
// Weighted cell-history partial sums (HBM-bound algorithmic floor, ~8.6 GB
// over the whole run). grid: (NS, B) blocks x 256 threads; slice s covers
// j in [s*64, min(t, s*64+64)). c_hist rows are streamed through LDS with
// double-buffered CDNA5 async global->LDS loads (ASYNCcnt in-order).
// Cpart[s][b][h] written unconditionally -> deterministic (no atomics).
// ---------------------------------------------------------------------------
__global__ void csum_kernel(const float* __restrict__ alpha,
                            const float* __restrict__ c_hist,
                            float* __restrict__ Cpart,
                            int t) {
    __shared__ float s_alpha[Td / NS];
    __shared__ float lbuf[2][CHUNK_J * Hd];
    int s   = blockIdx.x;
    int b   = blockIdx.y;
    int tid = threadIdx.x;
    int j0  = s * (Td / NS);
    int j1  = min(t, j0 + (Td / NS));

    if (tid < (Td / NS)) {
        int j = j0 + tid;
        s_alpha[tid] = (j < j1) ? alpha[(size_t)b * Td + j] : 0.0f;
    }

    const float* chb = c_hist + (size_t)b * Td * Hd;
    int nchunk = (j1 > j0) ? (j1 - j0 + CHUNK_J - 1) / CHUNK_J : 0;

    // each chunk = CHUNK_J*Hd floats = 1024 float4 slots; 4 slots per thread
    auto issue_chunk = [&](int jbase, float* dst) {
#pragma unroll
        for (int k = 0; k < 4; ++k) {
            int slot = tid + k * 256;
            int r    = slot >> 7;      // Hd/4 = 128 float4 per row
            int c4   = slot & 127;
            int j    = jbase + r;
            if (j >= j1) j = j1 - 1;   // clamp: data unused but keeps counts uniform
            const float* g = chb + (size_t)j * Hd + c4 * 4;
#if defined(USE_ASYNC_LDS)
            __builtin_amdgcn_global_load_async_to_lds_b128(
                (g_v4i*)g, (l_v4i*)(dst + slot * 4), 0, 0);
#else
            *(float4*)(dst + slot * 4) = *(const float4*)g;
#endif
        }
    };

    float a0 = 0.0f, a1 = 0.0f;
    if (nchunk > 0) issue_chunk(j0, lbuf[0]);
    for (int c = 0; c < nchunk; ++c) {
        if (c + 1 < nchunk) {
            issue_chunk(j0 + (c + 1) * CHUNK_J, lbuf[(c + 1) & 1]);
#if defined(USE_ASYNC_LDS)
            __builtin_amdgcn_s_wait_asynccnt(4); // chunk c done (in-order)
#endif
        } else {
#if defined(USE_ASYNC_LDS)
            __builtin_amdgcn_s_wait_asynccnt(0);
#endif
        }
        __syncthreads();
        const float* buf = lbuf[c & 1];
        int rmax = j1 - (j0 + c * CHUNK_J);
        if (rmax > CHUNK_J) rmax = CHUNK_J;
        for (int r = 0; r < rmax; ++r) {
            float a = s_alpha[c * CHUNK_J + r];
            a0 += a * buf[r * Hd + tid];
            a1 += a * buf[r * Hd + tid + 256];
        }
        __syncthreads(); // buffer reuse fence
    }
    Cpart[((size_t)s * Bd + b) * Hd + tid]       = a0;
    Cpart[((size_t)s * Bd + b) * Hd + tid + 256] = a1;
}

// ---------------------------------------------------------------------------
// One LSTM step: gates = v_t @ W_ih^T + h_prev @ W_hh^T + bias  (WMMA bf16),
// C = sum of NS Cpart slices, then c_new = sig(f)*C + sig(i)*tanh(g),
// h_new = sig(o)*tanh(c_new). Each wave owns one 16x16 tile in h-space and
// computes all 4 gate tiles. grid: 16 blocks x 128 threads (64 waves)
// ---------------------------------------------------------------------------
__global__ void lstm_step_kernel(const __bf16* __restrict__ v_bf,
                                 const __bf16* __restrict__ wih,
                                 const __bf16* __restrict__ whh,
                                 const float*  __restrict__ bias,
                                 const float*  __restrict__ Cpart,
                                 const __bf16* __restrict__ hprev,
                                 __bf16* __restrict__ hnext,
                                 float*  __restrict__ c_hist,
                                 float*  __restrict__ out,
                                 int t) {
    int wid  = blockIdx.x * (blockDim.x >> 5) + (threadIdx.x >> 5);
    int lane = threadIdx.x & 31;
    int l    = lane & 15;
    int hi   = lane >> 4;
    int tile_n = wid >> 1;       // 0..31  (h columns / 16)
    int tile_m = wid & 1;        // 0..1   (batch rows / 16)
    int n0 = tile_n * 16;
    int m0 = tile_m * 16;

    F8 acc[4];
#pragma unroll
    for (int g = 0; g < 4; ++g)
#pragma unroll
        for (int e = 0; e < 8; ++e) acc[g].f[e] = 0.0f;

    int arow = m0 + l; // batch index for this lane's A rows
    const __bf16* aptr_x = v_bf  + ((size_t)arow * Td + t) * Id + hi * 8;
    const __bf16* aptr_h = hprev + (size_t)arow * Hd + hi * 8;

    // x contribution: K = 128
#pragma unroll
    for (int k0 = 0; k0 < Id; k0 += 32) {
        bf16x16 a = load_frag(aptr_x + k0);
#pragma unroll
        for (int g = 0; g < 4; ++g) {
            const __bf16* bp = wih + ((size_t)(g * Hd + n0 + l)) * Id + k0 + hi * 8;
            bf16x16 b = load_frag(bp);
            acc[g].v = __builtin_amdgcn_wmma_f32_16x16x32_bf16(
                false, a, false, b, (short)0, acc[g].v, false, false);
        }
    }
    // h contribution: K = 512
    for (int k0 = 0; k0 < Hd; k0 += 32) {
        bf16x16 a = load_frag(aptr_h + k0);
#pragma unroll
        for (int g = 0; g < 4; ++g) {
            const __bf16* bp = whh + ((size_t)(g * Hd + n0 + l)) * Hd + k0 + hi * 8;
            bf16x16 b = load_frag(bp);
            acc[g].v = __builtin_amdgcn_wmma_f32_16x16x32_bf16(
                false, a, false, b, (short)0, acc[g].v, false, false);
        }
    }

    int   n  = n0 + l;
    float bi = bias[0 * Hd + n];
    float bf = bias[1 * Hd + n];
    float bg = bias[2 * Hd + n];
    float bo = bias[3 * Hd + n];
#pragma unroll
    for (int e = 0; e < 8; ++e) {
        int   m  = m0 + hi * 8 + e;             // C/D layout: lanes 16-31 hold M=8..15
        float gi = sigmoidf_(acc[0].f[e] + bi);
        float gf = sigmoidf_(acc[1].f[e] + bf);
        float gg = tanhf(acc[2].f[e] + bg);
        float go = sigmoidf_(acc[3].f[e] + bo);
        float cp = 0.0f;
#pragma unroll
        for (int s = 0; s < NS; ++s)
            cp += Cpart[((size_t)s * Bd + m) * Hd + n];
        float cn = gf * cp + gi * gg;
        float hn = go * tanhf(cn);
        size_t o = ((size_t)m * Td + t) * Hd + n;
        c_hist[o] = cn;
        out[o]    = hn;
        hnext[(size_t)m * Hd + n] = (__bf16)hn;
    }
}

// ---------------------------------------------------------------------------
extern "C" void kernel_launch(void* const* d_in, const int* in_sizes, int n_in,
                              void* d_out, int out_size, void* d_ws, size_t ws_size,
                              hipStream_t stream) {
    const float* values  = (const float*)d_in[0];
    const float* Deltas  = (const float*)d_in[1];
    /* ys (d_in[2]) is unused by the reference */
    const float* W_ih    = (const float*)d_in[3];
    const float* W_hh    = (const float*)d_in[4];
    const float* b_ih    = (const float*)d_in[5];
    const float* b_hh    = (const float*)d_in[6];
    const float* atten_W = (const float*)d_in[7];
    const float* atten_b = (const float*)d_in[8];
    float* out = (float*)d_out;

    // carve workspace (256B-aligned partitions)
    char* p = (char*)d_ws;
    auto carve = [&](size_t bytes) -> char* {
        char* r = p;
        p += (bytes + 255) & ~(size_t)255;
        return r;
    };
    float*  dcum      = (float*) carve((size_t)Bd * Td * 4);
    float*  vlin      = (float*) carve((size_t)Bd * Td * Id * 4);
    __bf16* vlin_bf   = (__bf16*)carve((size_t)Bd * Td * Id * 2);
    __bf16* values_bf = (__bf16*)carve((size_t)Bd * Td * Id * 2);
    __bf16* aw_bf     = (__bf16*)carve((size_t)Id * Id * 2);
    __bf16* wih_bf    = (__bf16*)carve((size_t)G4H * Id * 2);
    __bf16* whh_bf    = (__bf16*)carve((size_t)G4H * Hd * 2);
    float*  bias_sum  = (float*) carve((size_t)G4H * 4);
    float*  alpha     = (float*) carve((size_t)Bd * Td * 4);
    float*  Cpart     = (float*) carve((size_t)NS * Bd * Hd * 4);
    __bf16* hbuf0     = (__bf16*)carve((size_t)Bd * Hd * 2);
    __bf16* hbuf1     = (__bf16*)carve((size_t)Bd * Hd * 2);
    float*  c_hist    = (float*) carve((size_t)Bd * Td * Hd * 4);
    __bf16* hbuf[2]   = { hbuf0, hbuf1 };

    prep_kernel<<<512, 256, 0, stream>>>(values, W_ih, W_hh, b_ih, b_hh, atten_W,
                                         values_bf, wih_bf, whh_bf, aw_bf,
                                         bias_sum, Cpart, hbuf0);
    cumsum_kernel<<<1, 32, 0, stream>>>(Deltas, dcum);
    atten_gemm_kernel<<<(Bd * Td) / 16, 256, 0, stream>>>(values_bf, aw_bf, atten_b,
                                                          vlin, vlin_bf);
    // t = 0 : h_prev = 0 (hbuf0 zeroed), C = 0 (Cpart zeroed)
    lstm_step_kernel<<<16, 128, 0, stream>>>(vlin_bf, wih_bf, whh_bf, bias_sum,
                                             Cpart, hbuf[0], hbuf[1],
                                             c_hist, out, 0);
    for (int t = 1; t < Td; ++t) {
        alpha_kernel<<<Bd, 256, 0, stream>>>(values, vlin, dcum, alpha, t);
        csum_kernel<<<dim3(NS, Bd), 256, 0, stream>>>(alpha, c_hist, Cpart, t);
        lstm_step_kernel<<<16, 128, 0, stream>>>(vlin_bf, wih_bf, whh_bf, bias_sum,
                                                 Cpart, hbuf[t & 1], hbuf[(t + 1) & 1],
                                                 c_hist, out, t);
    }
}
